// TransformerVLM_89300960019209
// MI455X (gfx1250) — compile-verified, matrix operand order checked by reading, not run on hardware
//
#include <hip/hip_runtime.h>

// ---------------------------------------------------------------------------
// CDNA5 (gfx1250) transformer-VLM forward, wave32 + v_wmma_f32_16x16x32_bf16.
// Register-blocked GEMM: each wave owns a 64x32 output tile (4x2 WMMA frags,
// 8 accumulators), 4 waves/block -> 128x64 block tile. All B operands are
// pre-transposed to Bt[N,K] so fragments load as contiguous b128 pairs.
// ---------------------------------------------------------------------------

typedef __bf16 bf16_t;
typedef __attribute__((ext_vector_type(16))) __bf16 v16bf;
typedef __attribute__((ext_vector_type(8)))  float  v8f;

constexpr int B_ = 4, TTXT = 580, C_ = 1024, H_ = 16, L_ = 6, V_ = 16000;
constexpr int IMG = 336, PATCH = 16, NP = 441;       // 21*21 patches
constexpr int T_ = 1 + 1 + NP + 1 + TTXT;            // 1024
constexpr int HD = C_ / H_;                          // 64
constexpr int FF = 4 * C_;                           // 4096
constexpr int BOS = 1, IMG_START = 8, IMG_END = 9;
constexpr int KPATCH = 3 * PATCH * PATCH;            // 768
constexpr int MPATCH = B_ * NP;                      // 1764
constexpr int MPATCH_PAD = 1792;                     // 14 * 128

__device__ __forceinline__ bf16_t f2bf(float f) {
  unsigned u = __builtin_bit_cast(unsigned, f);
  unsigned r = u + 0x7FFFu + ((u >> 16) & 1u);       // round-to-nearest-even
  return __builtin_bit_cast(bf16_t, (unsigned short)(r >> 16));
}

__device__ __forceinline__ float wave_sum(float v) {
#pragma unroll
  for (int o = 16; o > 0; o >>= 1) v += __shfl_xor(v, o, 32);
  return v;
}
__device__ __forceinline__ float wave_max(float v) {
#pragma unroll
  for (int o = 16; o > 0; o >>= 1) v = fmaxf(v, __shfl_xor(v, o, 32));
  return v;
}

// Load one 16x32 bf16 fragment (ISA A/Bt layout): lane m = L&15 picks the row,
// K-halves split at lane 16; two contiguous 16B chunks per lane.
__device__ __forceinline__ v16bf load_frag(const bf16_t* __restrict__ p) {
  v16bf f;
#pragma unroll
  for (int i = 0; i < 8; ++i) f[i] = p[i];
#pragma unroll
  for (int i = 0; i < 8; ++i) f[8 + i] = p[16 + i];
  return f;
}

// ---------------------------------------------------------------------------
// Batched WMMA GEMM: D[M,N] = op(A[M,K] @ Bt[N,K]^T * scale + bias)(+D if ADD)
// grid = (N/64, M/128, batches); block = 128 threads (4 waves, 2x2 tile grid).
// Each wave: 64(M) x 32(N) via 4x2 fragments, 8 WMMAs per 32-wide K step.
// ---------------------------------------------------------------------------
template<bool ADD, bool RELU, bool OUT_BF16>
__global__ __launch_bounds__(128)
void gemm_wmma(const bf16_t* __restrict__ A, long lda,
               const bf16_t* __restrict__ Bt, long ldb,
               void* __restrict__ Out, long ldc,
               const float* __restrict__ bias,
               int M, int N, int K, float scale, int z2n,
               long sA1, long sA2, long sB1, long sB2, long sC1, long sC2)
{
  const int lane = threadIdx.x & 31;
  const int wave = threadIdx.x >> 5;
  const int wm = wave & 1;                 // wave row (0..1)
  const int wn = wave >> 1;                // wave col (0..1)

  const int z  = blockIdx.z;
  const int z1 = z / z2n, z2 = z % z2n;
  A  += (long)z1 * sA1 + (long)z2 * sA2;
  Bt += (long)z1 * sB1 + (long)z2 * sB2;
  const long outOff = (long)z1 * sC1 + (long)z2 * sC2;

  const int m0 = blockIdx.y * 128 + wm * 64;
  const int n0 = blockIdx.x * 64 + wn * 32;
  const int mr = lane & 15;                // row-in-tile (A) / col-in-tile (Bt)
  const int kh = (lane >> 4) * 8;          // K sub-offset per ISA layout

  const bf16_t* ap[4];
#pragma unroll
  for (int mi = 0; mi < 4; ++mi)
    ap[mi] = A + (long)(m0 + mi * 16 + mr) * lda + kh;
  const bf16_t* bp[2];
#pragma unroll
  for (int ni = 0; ni < 2; ++ni)
    bp[ni] = Bt + (long)(n0 + ni * 16 + mr) * ldb + kh;

  v8f acc[4][2];
#pragma unroll
  for (int mi = 0; mi < 4; ++mi)
#pragma unroll
    for (int ni = 0; ni < 2; ++ni) acc[mi][ni] = v8f{};

  for (int k0 = 0; k0 < K; k0 += 32) {
    v16bf af[4], bfr[2];
#pragma unroll
    for (int mi = 0; mi < 4; ++mi) {
      af[mi] = load_frag(ap[mi] + k0);
      __builtin_prefetch(ap[mi] + k0 + 32, 0, 1);   // global_prefetch_b8 (speculative)
    }
#pragma unroll
    for (int ni = 0; ni < 2; ++ni) {
      bfr[ni] = load_frag(bp[ni] + k0);
      __builtin_prefetch(bp[ni] + k0 + 32, 0, 1);
    }
#pragma unroll
    for (int mi = 0; mi < 4; ++mi)
#pragma unroll
      for (int ni = 0; ni < 2; ++ni)
        acc[mi][ni] = __builtin_amdgcn_wmma_f32_16x16x32_bf16(
            false, af[mi], false, bfr[ni], (short)0, acc[mi][ni], false, false);
  }

#pragma unroll
  for (int ni = 0; ni < 2; ++ni) {
    const int col = n0 + ni * 16 + mr;
    const float bv = bias ? bias[col] : 0.0f;
#pragma unroll
    for (int mi = 0; mi < 4; ++mi) {
#pragma unroll
      for (int r = 0; r < 8; ++r) {
        const int row = m0 + mi * 16 + r + (lane >> 4) * 8;   // ISA C/D layout
        if (row >= M) continue;
        float v = acc[mi][ni][r] * scale + bv;
        if constexpr (RELU) v = fmaxf(v, 0.0f);
        if constexpr (OUT_BF16) {
          bf16_t* o = (bf16_t*)Out + outOff;
          o[(long)row * ldc + col] = f2bf(v);
        } else {
          float* o = (float*)Out + outOff;
          if constexpr (ADD) v += o[(long)row * ldc + col];
          o[(long)row * ldc + col] = v;
        }
      }
    }
  }
}

// ------------------------------ elementwise --------------------------------
// fp32 [l][K][N] -> bf16 transposed [l][N][K]
__global__ void cvt_t_kernel(const float* __restrict__ in, bf16_t* __restrict__ out,
                             int K, int N, long total)
{
  long i = (long)blockIdx.x * blockDim.x + threadIdx.x;
  const long stride = (long)gridDim.x * blockDim.x;
  const long KN = (long)K * N;
  for (; i < total; i += stride) {
    const long l = i / KN;
    const long rem = i - l * KN;
    const int k = (int)(rem / N), n = (int)(rem % N);
    out[l * KN + (long)n * K + k] = f2bf(in[i]);
  }
}

__global__ void cvt_kernel(const float* __restrict__ in, bf16_t* __restrict__ out, long n)
{
  long i = (long)blockIdx.x * blockDim.x + threadIdx.x;
  const long stride = (long)gridDim.x * blockDim.x;
  for (; i < n; i += stride) out[i] = f2bf(in[i]);
}

// V [B,T,C] bf16 -> Vt [B,H,HD,T] bf16 (so P@V runs on the direct Bt path)
__global__ void vtrans_kernel(const bf16_t* __restrict__ v, bf16_t* __restrict__ vt)
{
  long i = (long)blockIdx.x * blockDim.x + threadIdx.x;
  if (i >= (long)B_ * T_ * C_) return;
  const int c = (int)(i % C_);
  const int t = (int)((i / C_) % T_);
  const int b = (int)(i / ((long)C_ * T_));
  const int hh = c / HD, d = c % HD;
  vt[(((long)b * H_ + hh) * HD + d) * T_ + t] = v[i];
}

__global__ void im2col_kernel(const float* __restrict__ img, bf16_t* __restrict__ Apat)
{
  long i = (long)blockIdx.x * blockDim.x + threadIdx.x;
  const long total = (long)MPATCH_PAD * KPATCH;
  if (i >= total) return;
  const int k = (int)(i % KPATCH);
  const int r = (int)(i / KPATCH);
  if (r >= MPATCH) { Apat[i] = f2bf(0.0f); return; }  // zero the pad rows
  const int b  = r / NP, p = r % NP;
  const int ci = k / (PATCH * PATCH);
  const int rem = k % (PATCH * PATCH);
  const int khh = rem / PATCH, kw = rem % PATCH;
  const int ph = p / 21, pw = p % 21;
  const float v = img[(((long)b * 3 + ci) * IMG + ph * PATCH + khh) * IMG + pw * PATCH + kw];
  Apat[i] = f2bf(v);
}

__global__ void embed_kernel(const int* __restrict__ tt,
                             const float* __restrict__ tok_emb,
                             const float* __restrict__ txt_pos,
                             const float* __restrict__ img_pos,
                             const float* __restrict__ patch_out,
                             float* __restrict__ x)
{
  long i = (long)blockIdx.x * blockDim.x + threadIdx.x;
  if (i >= (long)B_ * T_ * C_) return;
  const int c = (int)(i % C_);
  const int t = (int)((i / C_) % T_);
  const int b = (int)(i / ((long)C_ * T_));
  float v;
  if (t == 0)                 v = tok_emb[(long)BOS * C_ + c];
  else if (t == 1)            v = tok_emb[(long)IMG_START * C_ + c];
  else if (t < 2 + NP) {
    const int p = t - 2;
    v = patch_out[((long)b * NP + p) * C_ + c] + img_pos[(long)p * C_ + c];
  }
  else if (t == 2 + NP)       v = tok_emb[(long)IMG_END * C_ + c];
  else {
    const int tok = tt[(long)b * TTXT + (t - (3 + NP))];
    v = tok_emb[(long)tok * C_ + c];
  }
  x[i] = v + txt_pos[(long)t * C_ + c];
}

// LayerNorm over C=1024, output bf16 (GEMM A operand). One block per row.
__global__ __launch_bounds__(256)
void ln_kernel(const float* __restrict__ x, const float* __restrict__ g,
               const float* __restrict__ bta, bf16_t* __restrict__ out)
{
  __shared__ float ws[8], ws2[8];
  __shared__ float s_mean, s_rstd;
  const long row = blockIdx.x;
  const float* xr = x + row * C_;
  float s = 0.f, s2 = 0.f;
  for (int c = threadIdx.x; c < C_; c += 256) { const float v = xr[c]; s += v; s2 += v * v; }
  s = wave_sum(s); s2 = wave_sum(s2);
  if ((threadIdx.x & 31) == 0) { ws[threadIdx.x >> 5] = s; ws2[threadIdx.x >> 5] = s2; }
  __syncthreads();
  if (threadIdx.x == 0) {
    float ts = 0.f, t2 = 0.f;
#pragma unroll
    for (int i = 0; i < 8; ++i) { ts += ws[i]; t2 += ws2[i]; }
    const float m = ts / C_;
    s_mean = m;
    s_rstd = rsqrtf(t2 / C_ - m * m + 1e-5f);
  }
  __syncthreads();
  const float m = s_mean, r = s_rstd;
  for (int c = threadIdx.x; c < C_; c += 256)
    out[row * C_ + c] = f2bf((xr[c] - m) * r * g[c] + bta[c]);
}

// Causal softmax over a length-T row of f32 scores -> bf16 probs.
__global__ __launch_bounds__(256)
void softmax_kernel(const float* __restrict__ S, bf16_t* __restrict__ P)
{
  __shared__ float red[8];
  __shared__ float s_b;
  const long row = blockIdx.x;                        // over B*H*T
  const int  q   = (int)(row % T_);
  const float* sr = S + row * T_;
  bf16_t*      pr = P + row * T_;

  float mx = -3.0e38f;
  for (int k = threadIdx.x; k <= q; k += 256) mx = fmaxf(mx, sr[k]);
  mx = wave_max(mx);
  if ((threadIdx.x & 31) == 0) red[threadIdx.x >> 5] = mx;
  __syncthreads();
  if (threadIdx.x == 0) {
    float m = red[0];
#pragma unroll
    for (int i = 1; i < 8; ++i) m = fmaxf(m, red[i]);
    s_b = m;
  }
  __syncthreads();
  mx = s_b;

  float sum = 0.f;
  for (int k = threadIdx.x; k <= q; k += 256) sum += __expf(sr[k] - mx);
  sum = wave_sum(sum);
  __syncthreads();
  if ((threadIdx.x & 31) == 0) red[threadIdx.x >> 5] = sum;
  __syncthreads();
  if (threadIdx.x == 0) {
    float t = 0.f;
#pragma unroll
    for (int i = 0; i < 8; ++i) t += red[i];
    s_b = 1.0f / t;
  }
  __syncthreads();
  const float inv = s_b;
  for (int k = threadIdx.x; k < T_; k += 256)
    pr[k] = f2bf(k <= q ? __expf(sr[k] - mx) * inv : 0.0f);
}

// ---------------------------------------------------------------------------
extern "C" void kernel_launch(void* const* d_in, const int* in_sizes, int n_in,
                              void* d_out, int out_size, void* d_ws, size_t ws_size,
                              hipStream_t stream)
{
  (void)in_sizes; (void)n_in; (void)out_size; (void)ws_size;

  const int*   text_tokens = (const int*)  d_in[0];
  const float* images  = (const float*)d_in[1];
  const float* tok_emb = (const float*)d_in[2];
  const float* txt_pos = (const float*)d_in[3];
  const float* img_pos = (const float*)d_in[4];
  const float* patch_w = (const float*)d_in[5];
  const float* patch_b = (const float*)d_in[6];
  const float* ln1_g   = (const float*)d_in[7];
  const float* ln1_b   = (const float*)d_in[8];
  const float* wq      = (const float*)d_in[9];
  const float* wk      = (const float*)d_in[10];
  const float* wv      = (const float*)d_in[11];
  const float* wo      = (const float*)d_in[12];
  const float* bo      = (const float*)d_in[13];
  const float* ln2_g   = (const float*)d_in[14];
  const float* ln2_b   = (const float*)d_in[15];
  const float* w1      = (const float*)d_in[16];
  const float* b1      = (const float*)d_in[17];
  const float* w2      = (const float*)d_in[18];
  const float* b2      = (const float*)d_in[19];
  const float* lnf_g   = (const float*)d_in[20];
  const float* lnf_b   = (const float*)d_in[21];
  const float* lm_w    = (const float*)d_in[22];
  const float* lm_b    = (const float*)d_in[23];
  float* out = (float*)d_out;

  // ----- workspace carve-up (~650 MB) -----
  size_t off = 0;
  auto alloc = [&](size_t bytes) -> void* {
    void* p = (char*)d_ws + off;
    off += (bytes + 255) & ~(size_t)255;
    return p;
  };
  bf16_t* wq_t  = (bf16_t*)alloc((size_t)L_ * C_ * C_ * 2);   // [l][N=C][K=C]
  bf16_t* wk_t  = (bf16_t*)alloc((size_t)L_ * C_ * C_ * 2);
  bf16_t* wv_t  = (bf16_t*)alloc((size_t)L_ * C_ * C_ * 2);
  bf16_t* wo_t  = (bf16_t*)alloc((size_t)L_ * C_ * C_ * 2);
  bf16_t* w1_t  = (bf16_t*)alloc((size_t)L_ * C_ * FF * 2);   // [l][FF][C]
  bf16_t* w2_t  = (bf16_t*)alloc((size_t)L_ * FF * C_ * 2);   // [l][C][FF]
  bf16_t* lmw_t = (bf16_t*)alloc((size_t)C_ * V_ * 2);        // [V][C]
  bf16_t* pw_b  = (bf16_t*)alloc((size_t)C_ * KPATCH * 2);    // already Bt form
  bf16_t* Apat  = (bf16_t*)alloc((size_t)MPATCH_PAD * KPATCH * 2);
  float*  pout  = (float*) alloc((size_t)MPATCH * C_ * 4);
  float*  x     = (float*) alloc((size_t)B_ * T_ * C_ * 4);
  bf16_t* h     = (bf16_t*)alloc((size_t)B_ * T_ * C_ * 2);
  bf16_t* qb    = (bf16_t*)alloc((size_t)B_ * T_ * C_ * 2);
  bf16_t* kb    = (bf16_t*)alloc((size_t)B_ * T_ * C_ * 2);
  bf16_t* vb    = (bf16_t*)alloc((size_t)B_ * T_ * C_ * 2);
  bf16_t* vt    = (bf16_t*)alloc((size_t)B_ * T_ * C_ * 2);   // [B,H,HD,T]
  float*  S     = (float*) alloc((size_t)B_ * H_ * T_ * T_ * 4);
  bf16_t* P     = (bf16_t*)alloc((size_t)B_ * H_ * T_ * T_ * 2);
  bf16_t* aO    = (bf16_t*)alloc((size_t)B_ * T_ * C_ * 2);
  bf16_t* ffb   = (bf16_t*)alloc((size_t)B_ * T_ * FF * 2);

  const dim3 blk(128, 1, 1);
  const int  BT = B_ * T_;                            // 4096

  // ----- weights fp32 -> bf16, transposed to Bt[N,K] -----
  cvt_t_kernel<<<4096, 256, 0, stream>>>(wq,   wq_t,  C_, C_, (long)L_ * C_ * C_);
  cvt_t_kernel<<<4096, 256, 0, stream>>>(wk,   wk_t,  C_, C_, (long)L_ * C_ * C_);
  cvt_t_kernel<<<4096, 256, 0, stream>>>(wv,   wv_t,  C_, C_, (long)L_ * C_ * C_);
  cvt_t_kernel<<<4096, 256, 0, stream>>>(wo,   wo_t,  C_, C_, (long)L_ * C_ * C_);
  cvt_t_kernel<<<4096, 256, 0, stream>>>(w1,   w1_t,  C_, FF, (long)L_ * C_ * FF);
  cvt_t_kernel<<<4096, 256, 0, stream>>>(w2,   w2_t,  FF, C_, (long)L_ * FF * C_);
  cvt_t_kernel<<<4096, 256, 0, stream>>>(lm_w, lmw_t, C_, V_, (long)C_ * V_);
  cvt_kernel  <<<4096, 256, 0, stream>>>(patch_w, pw_b, (long)C_ * KPATCH);

  // ----- patch embedding: im2col + GEMM (Bt = patch_w [C,768]) -----
  {
    const long total = (long)MPATCH_PAD * KPATCH;
    im2col_kernel<<<(unsigned)((total + 255) / 256), 256, 0, stream>>>(images, Apat);
    gemm_wmma<false, false, false>
      <<<dim3(C_ / 64, MPATCH_PAD / 128, 1), blk, 0, stream>>>(
        Apat, KPATCH, pw_b, KPATCH, pout, C_, patch_b,
        MPATCH, C_, KPATCH, 1.0f, 1, 0, 0, 0, 0, 0, 0);
  }

  // ----- token/patch/pos assembly -----
  embed_kernel<<<(unsigned)(((long)BT * C_ + 255) / 256), 256, 0, stream>>>(
      text_tokens, tok_emb, txt_pos, img_pos, pout, x);

  // ----- transformer blocks -----
  for (int l = 0; l < L_; ++l) {
    const bf16_t* Wq = wq_t + (size_t)l * C_ * C_;
    const bf16_t* Wk = wk_t + (size_t)l * C_ * C_;
    const bf16_t* Wv = wv_t + (size_t)l * C_ * C_;
    const bf16_t* Wo = wo_t + (size_t)l * C_ * C_;
    const bf16_t* W1 = w1_t + (size_t)l * C_ * FF;
    const bf16_t* W2 = w2_t + (size_t)l * FF * C_;

    ln_kernel<<<BT, 256, 0, stream>>>(x, ln1_g + (size_t)l * C_, ln1_b + (size_t)l * C_, h);

    gemm_wmma<false, false, true><<<dim3(C_ / 64, BT / 128, 1), blk, 0, stream>>>(
        h, C_, Wq, C_, qb, C_, nullptr, BT, C_, C_, 1.0f, 1, 0, 0, 0, 0, 0, 0);
    gemm_wmma<false, false, true><<<dim3(C_ / 64, BT / 128, 1), blk, 0, stream>>>(
        h, C_, Wk, C_, kb, C_, nullptr, BT, C_, C_, 1.0f, 1, 0, 0, 0, 0, 0, 0);
    gemm_wmma<false, false, true><<<dim3(C_ / 64, BT / 128, 1), blk, 0, stream>>>(
        h, C_, Wv, C_, vb, C_, nullptr, BT, C_, C_, 1.0f, 1, 0, 0, 0, 0, 0, 0);

    vtrans_kernel<<<(unsigned)(((long)BT * C_ + 255) / 256), 256, 0, stream>>>(vb, vt);

    // S[b,h] = (Q_bh @ K_bh^T) * HD^-0.5   (batched over B*H via grid.z)
    gemm_wmma<false, false, false><<<dim3(T_ / 64, T_ / 128, B_ * H_), blk, 0, stream>>>(
        qb, C_, kb, C_, S, T_, nullptr, T_, T_, HD, 0.125f, H_,
        (long)T_ * C_, (long)HD, (long)T_ * C_, (long)HD,
        (long)H_ * T_ * T_, (long)T_ * T_);

    softmax_kernel<<<B_ * H_ * T_, 256, 0, stream>>>(S, P);

    // O[b,:,h,:] = P[b,h] @ V_bh   (Bt = Vt[b,h] : [HD, T])
    gemm_wmma<false, false, true><<<dim3(HD / 64, T_ / 128, B_ * H_), blk, 0, stream>>>(
        P, T_, vt, T_, aO, C_, nullptr, T_, HD, T_, 1.0f, H_,
        (long)H_ * T_ * T_, (long)T_ * T_, (long)H_ * HD * T_, (long)HD * T_,
        (long)T_ * C_, (long)HD);

    // x += O @ Wo + bo
    gemm_wmma<true, false, false><<<dim3(C_ / 64, BT / 128, 1), blk, 0, stream>>>(
        aO, C_, Wo, C_, x, C_, bo + (size_t)l * C_, BT, C_, C_, 1.0f, 1, 0, 0, 0, 0, 0, 0);

    // MLP: x += relu(ln2(x) @ W1 + b1) @ W2 + b2
    ln_kernel<<<BT, 256, 0, stream>>>(x, ln2_g + (size_t)l * C_, ln2_b + (size_t)l * C_, h);
    gemm_wmma<false, true, true><<<dim3(FF / 64, BT / 128, 1), blk, 0, stream>>>(
        h, C_, W1, C_, ffb, FF, b1 + (size_t)l * FF, BT, FF, C_, 1.0f, 1, 0, 0, 0, 0, 0, 0);
    gemm_wmma<true, false, false><<<dim3(C_ / 64, BT / 128, 1), blk, 0, stream>>>(
        ffb, FF, W2, FF, x, C_, b2 + (size_t)l * C_, BT, C_, FF, 1.0f, 1, 0, 0, 0, 0, 0, 0);
  }

  // ----- final LN + LM head -----
  ln_kernel<<<BT, 256, 0, stream>>>(x, lnf_g, lnf_b, h);
  gemm_wmma<false, false, false><<<dim3(V_ / 64, BT / 128, 1), blk, 0, stream>>>(
      h, C_, lmw_t, C_, out, V_, lm_b, BT, V_, C_, 1.0f, 1, 0, 0, 0, 0, 0, 0);
}
